// scQDiffComm_56109452754997
// MI455X (gfx1250) — matrix operand order repaired and checked by reading, not compiled, not run on hardware
//
#include <hip/hip_runtime.h>

#define N_TOK 16384
#define D_DIM 64
#define H_DIM 256

typedef float  v8f  __attribute__((ext_vector_type(8)));
typedef __bf16 v16bf __attribute__((ext_vector_type(16)));
typedef unsigned short v16u  __attribute__((ext_vector_type(16)));
typedef unsigned int   v8u32 __attribute__((ext_vector_type(8)));
typedef unsigned int   v4u   __attribute__((ext_vector_type(4)));
typedef int            v8i_t __attribute__((ext_vector_type(8)));
typedef int            v4i_t __attribute__((ext_vector_type(4)));

// LDS (address_space(3)) views so B-fragment reads lower to ds_load_b128
typedef __attribute__((address_space(3))) unsigned short lds_u16;
typedef __attribute__((address_space(3))) const v16bf    lds_v16bf;

__device__ __forceinline__ unsigned short f32_to_bf16_rne(float f) {
    unsigned u = __builtin_bit_cast(unsigned, f);
    unsigned r = (u + 0x7fffu + ((u >> 16) & 1u)) >> 16;
    return (unsigned short)r;
}
__device__ __forceinline__ float bf16_to_f32(unsigned short h) {
    unsigned u = ((unsigned)h) << 16;
    return __builtin_bit_cast(float, u);
}

// ---------------------------------------------------------------------------
// Kernel 1: per row j
//   XB[j][d] = dot(X[j,:], B[d,:]); stored transposed + split bf16 (hi/lo)
//   u_intra[j,:] = tanh([X_j, t] @ W1 + b1) @ W2 + b2   -> d_out
// ---------------------------------------------------------------------------
__global__ __launch_bounds__(256)
void xb_intra_kernel(const float* __restrict__ X,
                     const float* __restrict__ t,
                     const float* __restrict__ B,
                     const float* __restrict__ W1,
                     const float* __restrict__ b1,
                     const float* __restrict__ W2,
                     const float* __restrict__ b2,
                     unsigned short* __restrict__ xbt_h,
                     unsigned short* __restrict__ xbt_l,
                     float* __restrict__ out) {
    const int j = blockIdx.x * blockDim.x + threadIdx.x;
    if (j >= N_TOK) return;

    float x[D_DIM];
#pragma unroll
    for (int k = 0; k < D_DIM; ++k) x[k] = X[(size_t)j * D_DIM + k];

    for (int d = 0; d < D_DIM; ++d) {
        float s = 0.f;
#pragma unroll
        for (int k = 0; k < D_DIM; ++k) s = fmaf(x[k], B[d * D_DIM + k], s);
        unsigned u = __builtin_bit_cast(unsigned, s);
        float hiF  = __builtin_bit_cast(float, u & 0xffff0000u);   // truncated hi
        xbt_h[(size_t)d * N_TOK + j] = (unsigned short)(u >> 16);
        xbt_l[(size_t)d * N_TOK + j] = f32_to_bf16_rne(s - hiF);   // residual
    }

    const float tv = t[0];
    float acc[D_DIM];
#pragma unroll
    for (int d = 0; d < D_DIM; ++d) acc[d] = b2[d];
    for (int hh = 0; hh < H_DIM; ++hh) {
        float s = b1[hh] + tv * W1[D_DIM * H_DIM + hh];
#pragma unroll
        for (int k = 0; k < D_DIM; ++k) s = fmaf(x[k], W1[k * H_DIM + hh], s);
        float hv = tanhf(s);
#pragma unroll
        for (int d = 0; d < D_DIM; ++d) acc[d] = fmaf(hv, W2[hh * D_DIM + d], acc[d]);
    }
#pragma unroll
    for (int d = 0; d < D_DIM; ++d) out[(size_t)j * D_DIM + d] = acc[d];
}

// ---------------------------------------------------------------------------
// TDM: load one 64(d) x 32(k) bf16 tile of XBT[64][16384] into LDS.
// 2D tensor: dim0 = k (contiguous, stride 1), dim1 = d (stride 16384).
// LDS result: row d at lds_off + d*64 bytes, 32 contiguous bf16 per row.
// ---------------------------------------------------------------------------
__device__ __forceinline__ void tdm_load_tile(unsigned lds_byte_off,
                                              const unsigned short* gptr) {
    unsigned long long ga = (unsigned long long)(uintptr_t)gptr;
    v4u g0;
    g0[0] = 1u;                                       // count=1, user descriptor
    g0[1] = lds_byte_off;                             // lds_addr
    g0[2] = (unsigned)(ga & 0xffffffffu);             // global_addr[31:0]
    g0[3] = (unsigned)((ga >> 32) & 0x01ffffffu) | (2u << 30);  // [56:32], type=2
    const unsigned td0 = 16384u, td1 = 64u, tile0 = 32u, tile1 = 64u;
    const unsigned long long t0s = 16384ull;          // tensor_dim0_stride
    v8i_t g1;
    g1[0] = (int)(1u << 16);                          // wg_mask=0, data_size=1 (2B)
    g1[1] = (int)((td0 & 0xffffu) << 16);             // [15:0] atomic_bar=0, [31:16]=td0.lo
    g1[2] = (int)((td0 >> 16) | ((td1 & 0xffffu) << 16));
    g1[3] = (int)((td1 >> 16) | (tile0 << 16));       // [31:16]=tile_dim0
    g1[4] = (int)(tile1);                             // [15:0]=tile_dim1, tile_dim2=0
    g1[5] = (int)(t0s & 0xffffffffull);               // stride0[31:0]
    g1[6] = (int)((t0s >> 32) & 0xffffull);           // stride0[47:32], stride1.lo=0
    g1[7] = 0;
    v4i_t z4 = {0, 0, 0, 0};
    v8i_t z8 = {0, 0, 0, 0, 0, 0, 0, 0};
    __builtin_amdgcn_tensor_load_to_lds(g0, g1, z4, z4, z8, 0);
}

// pack top-16-bits of two f32 into one dword: [lo]=u0>>16, [hi]=u1>>16
__device__ __forceinline__ unsigned pack_hi16(unsigned u0, unsigned u1) {
    return __builtin_amdgcn_perm(u1, u0, 0x07060302u);
}

// ---------------------------------------------------------------------------
// Kernel 2: u_comm = Wt @ XB - rowsum(Wt) * XB, fused into d_out (+= u_intra).
// Block = 4 waves; each wave one 16-row strip x all 64 cols -> Wt read once.
// B tiles (hi/lo) TDM-staged into double-buffered LDS, shared by all 4 waves.
// Split-precision bf16 WMMA: C += Ah*Bh + Ah*Bl + Al*Bh (f32 accumulate).
// rowsum(Wt) via WMMA against an all-ones B (lands directly in C layout).
// ---------------------------------------------------------------------------
__global__ __launch_bounds__(128, 1)
void comm_gemm_kernel(const float* __restrict__ Wt,
                      const unsigned short* __restrict__ xbt_h,
                      const unsigned short* __restrict__ xbt_l,
                      float* __restrict__ out) {
    __shared__ alignas(64) unsigned short ldsB[2][2][D_DIM * 32]; // 16 KB

    const int lane  = threadIdx.x & 31;
    const int wave  = __builtin_amdgcn_readfirstlane(threadIdx.x >> 5);
    const int strip = blockIdx.x * 4 + wave;            // 0..1023
    const int row0  = strip * 16;
    const int g     = lane >> 4;                        // A/B lane group
    const int r     = lane & 15;                        // A row / B column

    const float* arow = Wt + (size_t)(row0 + r) * N_TOK;

    v8f zero = {0.f, 0.f, 0.f, 0.f, 0.f, 0.f, 0.f, 0.f};
    v8f acc[4], acc_rs = zero;
#pragma unroll
    for (int c = 0; c < 4; ++c) acc[c] = zero;

    v16u onesu;
#pragma unroll
    for (int e = 0; e < 16; ++e) onesu[e] = 0x3f80u;    // bf16 1.0
    const v16bf bones = __builtin_bit_cast(v16bf, onesu);

    // prologue: stage k0=0 into buffer 0 (wave 0 only; TDM ignores EXEC)
    if (wave == 0) {
        tdm_load_tile(0u, xbt_h);                       // buf0 / hi at LDS 0
        tdm_load_tile(4096u, xbt_l);                    // buf0 / lo at LDS 4096
    }

    int buf = 0;
    for (int k0 = 0; k0 < N_TOK; k0 += 32) {
        if (wave == 0) __builtin_amdgcn_s_wait_tensorcnt(0);
        __syncthreads();                                // current buf ready for all

        if (k0 + 32 < N_TOK && wave == 0) {             // stage next chunk
            unsigned nb = (unsigned)(buf ^ 1) * 8192u;
            tdm_load_tile(nb,         xbt_h + (k0 + 32));
            tdm_load_tile(nb + 4096u, xbt_l + (k0 + 32));
        }

        // --- A: 16x32 f32 -> split bf16 hi/lo (layout: lane(g,r) holds row r,
        //     K = {16g..16g+15})
        v8f a0 = *(const v8f*)(arow + k0 + 16 * g);
        v8f a1 = *(const v8f*)(arow + k0 + 16 * g + 8);
        __builtin_prefetch(arow + k0 + 256, 0, 0);

        v8u32 hw, lw;
#pragma unroll
        for (int w = 0; w < 4; ++w) {
            float f0 = a0[2 * w], f1 = a0[2 * w + 1];
            float f2 = a1[2 * w], f3 = a1[2 * w + 1];
            unsigned u0 = __builtin_bit_cast(unsigned, f0);
            unsigned u1 = __builtin_bit_cast(unsigned, f1);
            unsigned u2 = __builtin_bit_cast(unsigned, f2);
            unsigned u3 = __builtin_bit_cast(unsigned, f3);
            hw[w]     = pack_hi16(u0, u1);
            hw[w + 4] = pack_hi16(u2, u3);
            float r0 = f0 - __builtin_bit_cast(float, u0 & 0xffff0000u);
            float r1 = f1 - __builtin_bit_cast(float, u1 & 0xffff0000u);
            float r2 = f2 - __builtin_bit_cast(float, u2 & 0xffff0000u);
            float r3 = f3 - __builtin_bit_cast(float, u3 & 0xffff0000u);
            unsigned t0 = __builtin_bit_cast(unsigned, r0);
            unsigned t1 = __builtin_bit_cast(unsigned, r1);
            unsigned t2 = __builtin_bit_cast(unsigned, r2);
            unsigned t3 = __builtin_bit_cast(unsigned, r3);
            t0 += 0x7fffu + ((t0 >> 16) & 1u);          // RNE in the top 16 bits
            t1 += 0x7fffu + ((t1 >> 16) & 1u);
            t2 += 0x7fffu + ((t2 >> 16) & 1u);
            t3 += 0x7fffu + ((t3 >> 16) & 1u);
            lw[w]     = pack_hi16(t0, t1);
            lw[w + 4] = pack_hi16(t2, t3);
        }
        v16bf ah = __builtin_bit_cast(v16bf, hw);
        v16bf al = __builtin_bit_cast(v16bf, lw);

        // rowsum(Wt) tile: WMMA against ones -> row sums in C layout
        acc_rs = __builtin_amdgcn_wmma_f32_16x16x32_bf16(
                     false, ah, false, bones, (short)0, acc_rs, false, false);
        acc_rs = __builtin_amdgcn_wmma_f32_16x16x32_bf16(
                     false, al, false, bones, (short)0, acc_rs, false, false);

        // --- B fragments from LDS (lane(g,r): column r of tile c, K=16g..16g+15)
        // Launder the LDS pointers: TDM wrote this memory behind the compiler's
        // back (address passed as integer in the descriptor), so force the
        // loads to be real and ordered after the barrier.
        lds_u16* lh = (lds_u16*)&ldsB[buf][0][0];
        lds_u16* ll = (lds_u16*)&ldsB[buf][1][0];
        asm volatile("" : "+v"(lh), "+v"(ll) :: "memory");
#pragma unroll
        for (int c = 0; c < 4; ++c) {
            int boff = (c * 16 + r) * 32 + 16 * g;
            v16bf bh = *(lds_v16bf*)(lh + boff);
            v16bf bl = *(lds_v16bf*)(ll + boff);
            acc[c] = __builtin_amdgcn_wmma_f32_16x16x32_bf16(
                         false, ah, false, bh, (short)0, acc[c], false, false);
            acc[c] = __builtin_amdgcn_wmma_f32_16x16x32_bf16(
                         false, ah, false, bl, (short)0, acc[c], false, false);
            acc[c] = __builtin_amdgcn_wmma_f32_16x16x32_bf16(
                         false, al, false, bh, (short)0, acc[c], false, false);
        }
        buf ^= 1;
    }

    // C layout: lanes0-15 N=lane, VGPR v -> M=v ; lanes16-31 N=lane-16, M=8+v.
    // acc_rs[v] = rowsum(Wt[row0+v+8g,:]) -- same layout, no shuffles needed.
#pragma unroll
    for (int c = 0; c < 4; ++c) {
        const int d = c * 16 + r;
#pragma unroll
        for (int v = 0; v < 8; ++v) {
            const int i = row0 + v + 8 * g;
            float xb = bf16_to_f32(xbt_h[(size_t)d * N_TOK + i]) +
                       bf16_to_f32(xbt_l[(size_t)d * N_TOK + i]);
            size_t o = (size_t)i * D_DIM + d;
            out[o] = out[o] + acc[c][v] - acc_rs[v] * xb;
        }
    }
}

extern "C" void kernel_launch(void* const* d_in, const int* in_sizes, int n_in,
                              void* d_out, int out_size, void* d_ws, size_t ws_size,
                              hipStream_t stream) {
    const float* X  = (const float*)d_in[0];
    const float* t  = (const float*)d_in[1];
    const float* Wt = (const float*)d_in[2];
    const float* B  = (const float*)d_in[3];
    const float* W1 = (const float*)d_in[4];
    const float* b1 = (const float*)d_in[5];
    const float* W2 = (const float*)d_in[6];
    const float* b2 = (const float*)d_in[7];
    float* out = (float*)d_out;

    unsigned short* xbt_h = (unsigned short*)d_ws;               // 2 MB
    unsigned short* xbt_l = xbt_h + (size_t)D_DIM * N_TOK;       // 2 MB

    hipLaunchKernelGGL(xb_intra_kernel, dim3(N_TOK / 256), dim3(256), 0, stream,
                       X, t, B, W1, b1, W2, b2, xbt_h, xbt_l, out);
    hipLaunchKernelGGL(comm_gemm_kernel, dim3(256), dim3(128), 0, stream,
                       Wt, xbt_h, xbt_l, out);
}
